// CausalSelfAttention_13331578487038
// MI455X (gfx1250) — compile-verified, matrix operand order checked by reading, not run on hardware
//
#include <hip/hip_runtime.h>
#include <hip/hip_bf16.h>

// CausalSelfAttention for MI455X (gfx1250, wave32, WMMA bf16, async->LDS).
// B=2, S=2048, D=1024, H=16, HD=64.

#define BB 2
#define SS 2048
#define DD 1024
#define HH 16
#define HD 64
#define D3 3072
#define BS (BB * SS)   // 4096

#define LDSP 40        // padded LDS row stride (bf16 elems): 80B = 20 banks -> conflict-free

typedef __bf16 bf16_t;
typedef __attribute__((ext_vector_type(16))) __bf16 v16bf;
typedef __attribute__((ext_vector_type(8)))  __bf16 v8bf;
typedef __attribute__((ext_vector_type(8)))  float  v8f;

// ---------------------------------------------------------------------------
// WMMA helper: D = A(16x32 bf16) * B(32x16 bf16) + C(16x16 f32)
// ---------------------------------------------------------------------------
__device__ __forceinline__ v8f wmma_bf16(v16bf a, v16bf b, v8f c) {
  return __builtin_amdgcn_wmma_f32_16x16x32_bf16(
      /*neg_a=*/false, a, /*neg_b=*/false, b,
      /*c_mod=*/(short)0, c, /*reuse_a=*/false, /*reuse_b=*/false);
}

// Fragment loader for a 16x32 tile from a row-major [16][ld] bf16 matrix.
// CDNA5 16-bit A-matrix layout (ISA 7.12.2):
//   lanes 0-15  : row = lane,    K = {0..7, 16..23}
//   lanes 16-31 : row = lane-16, K = {8..15, 24..31}
// Same per-lane pattern serves as the B operand when B is stored transposed
// ([N][K] row-major): lanes index N, registers index K.
__device__ __forceinline__ v16bf load_frag(const bf16_t* base, int ld) {
  const int lane = threadIdx.x & 31;
  const int r = lane & 15;
  const int hi = lane >> 4;
  const bf16_t* p = base + r * ld + hi * 8;
  v8bf lo = *(const v8bf*)(p);
  v8bf hh = *(const v8bf*)(p + 16);
  v16bf out;
#pragma unroll
  for (int i = 0; i < 8; ++i) { out[i] = lo[i]; out[i + 8] = hh[i]; }
  return out;
}

// ---------------------------------------------------------------------------
// Async global -> LDS stage of one 64(N) x 32(K) bf16 tile of Bt[N][K].
// 256 threads x one b128 each; rows padded to LDSP elems in LDS.
// Tracked by ASYNCcnt; drained with s_wait_asynccnt.
// ---------------------------------------------------------------------------
__device__ __forceinline__ void stage_b_tile(const bf16_t* __restrict__ Bt,
                                             int col0, int kp, bf16_t* dst) {
  const int tid = threadIdx.x & 255;
  const int n_ = tid >> 2;             // 0..63  (N row)
  const int ko = (tid & 3) * 8;        // 0,8,16,24 (K offset, 8 bf16 = 16B)
  const bf16_t* gp = Bt + (size_t)(col0 + n_) * DD + kp + ko;
  const bf16_t* lp = dst + n_ * LDSP + ko;
  unsigned loff = (unsigned)(size_t)lp;                 // low 32 bits = LDS addr
  unsigned long long ga = (unsigned long long)(size_t)gp;
  asm volatile("global_load_async_to_lds_b128 %0, %1, off"
               :: "v"(loff), "v"(ga) : "memory");
}

__device__ __forceinline__ void wait_async0() {
  asm volatile("s_wait_asynccnt 0x0" ::: "memory");
}

// ---------------------------------------------------------------------------
// Shared GEMM core: block tile 256(M) x 64(N), K = DD.
// 8 waves x 32 rows each; B tile staged once per block via async->LDS,
// double-buffered:
//   iter i: wait(async for buf) ; barrier ; stage buf^1 for i+1 ; compute buf.
// Per wave per K-step: 2 A frags (global), 4 B frags (LDS), 8 WMMAs.
// ---------------------------------------------------------------------------
__device__ __forceinline__ void gemm_core_256x64(
    const bf16_t* __restrict__ A, const bf16_t* __restrict__ Bt,
    int row0, int col0, bf16_t* smem /* [2][64*LDSP] */, v8f c[2][4]) {
  stage_b_tile(Bt, col0, 0, smem);
  v16bf a_cur0 = load_frag(A + (size_t)row0 * DD, DD);
  v16bf a_cur1 = load_frag(A + (size_t)(row0 + 16) * DD, DD);
  int buf = 0;
  for (int kp = 0; kp < DD; kp += 32) {
    wait_async0();        // buf tile resident (this wave's async loads done)
    __syncthreads();      // all waves staged buf; all done reading buf^1
    v16bf a_n0 = a_cur0, a_n1 = a_cur1;
    if (kp + 32 < DD) {
      stage_b_tile(Bt, col0, kp + 32, smem + (buf ^ 1) * (64 * LDSP));
      a_n0 = load_frag(A + (size_t)row0 * DD + kp + 32, DD);
      a_n1 = load_frag(A + (size_t)(row0 + 16) * DD + kp + 32, DD);
    }
    const bf16_t* bsm = smem + buf * (64 * LDSP);
    v16bf bfr[4];
#pragma unroll
    for (int t = 0; t < 4; ++t)
      bfr[t] = load_frag(bsm + (16 * t) * LDSP, LDSP);
#pragma unroll
    for (int t = 0; t < 4; ++t) {
      c[0][t] = wmma_bf16(a_cur0, bfr[t], c[0][t]);
      c[1][t] = wmma_bf16(a_cur1, bfr[t], c[1][t]);
    }
    a_cur0 = a_n0;
    a_cur1 = a_n1;
    buf ^= 1;
  }
}

// ---------------------------------------------------------------------------
// Stage 0a: f32 -> bf16 elementwise
// ---------------------------------------------------------------------------
__global__ void k_cvt_bf16(const float* __restrict__ in, bf16_t* __restrict__ out, int n) {
  int i = blockIdx.x * blockDim.x + threadIdx.x;
  if (i < n) out[i] = (bf16_t)in[i];
}

// Stage 0b: transpose-convert W [K][N] f32 -> Wt [N][K] bf16
__global__ void k_transpose_bf16(const float* __restrict__ in, bf16_t* __restrict__ out,
                                 int K, int N) {
  int i = blockIdx.x * blockDim.x + threadIdx.x;
  if (i < K * N) {
    int k = i / N, n = i % N;
    out[n * K + k] = (bf16_t)in[i];
  }
}

// ---------------------------------------------------------------------------
// Stage 1: QKV GEMM. C[4096,3072] = Xb @ Wqkv + bqkv, scattered to
// q[B,H,S,64], k[B,H,S,64], vT[B,H,64,S] in bf16.
// ---------------------------------------------------------------------------
__global__ __launch_bounds__(256) void k_gemm_qkv(
    const bf16_t* __restrict__ A, const bf16_t* __restrict__ Bt,
    const float* __restrict__ bias,
    bf16_t* __restrict__ q, bf16_t* __restrict__ kk_, bf16_t* __restrict__ vt) {
  __shared__ __align__(16) bf16_t smem[2 * 64 * LDSP];
  const int wave = threadIdx.x >> 5;
  const int lane = threadIdx.x & 31;
  const int row0 = blockIdx.x * 256 + wave * 32;
  const int col0 = blockIdx.y * 64;

  v8f c[2][4] = {};
  gemm_core_256x64(A, Bt, row0, col0, smem, c);

  const int r_ = lane & 15, hi = lane >> 4;
#pragma unroll
  for (int mi = 0; mi < 2; ++mi) {
#pragma unroll
    for (int t = 0; t < 4; ++t) {
#pragma unroll
      for (int r = 0; r < 8; ++r) {
        const int m = row0 + 16 * mi + r + hi * 8;  // token index in [0, 4096)
        const int n = col0 + 16 * t + r_;           // column in [0, 3072)
        const float v = c[mi][t][r] + bias[n];
        const int b_ = m >> 11, s = m & (SS - 1);
        const int seg = n >> 10, w = n & (DD - 1);
        const int h = w >> 6, hd = w & (HD - 1);
        const bf16_t bv = (bf16_t)v;
        if (seg == 0)      q  [(((b_ * HH + h) * SS) + s) * HD + hd] = bv;
        else if (seg == 1) kk_[(((b_ * HH + h) * SS) + s) * HD + hd] = bv;
        else               vt [(((b_ * HH + h) * HD) + hd) * SS + s] = bv;
      }
    }
  }
}

// ---------------------------------------------------------------------------
// Stage 2: flash attention. One wave per 16-query tile; online softmax over
// 32-key chunks. Scores/PV via WMMA; P staged through per-wave LDS to convert
// C-fragment layout into A-fragment layout.
// ---------------------------------------------------------------------------
#define NEGBIG (-1.0e30f)

__global__ __launch_bounds__(256) void k_attn(
    const bf16_t* __restrict__ Q, const bf16_t* __restrict__ K,
    const bf16_t* __restrict__ VT, const unsigned char* __restrict__ mask,
    bf16_t* __restrict__ outp) {
  __shared__ __align__(16) bf16_t plds[8][16 * LDSP];

  const int wave = threadIdx.x >> 5;
  const int lane = threadIdx.x & 31;
  const int g = blockIdx.x * 8 + wave;          // tile id in [0, 4096)
  const int q0 = (g & 127) << 4;                // query tile start
  const int h = (g >> 7) & (HH - 1);
  const int b = g >> 11;

  const bf16_t* qp = Q + ((b * HH + h) * SS) * HD;
  const bf16_t* kp = K + ((b * HH + h) * SS) * HD;
  const bf16_t* vp = VT + ((b * HH + h) * HD) * SS;
  const unsigned char* mk = mask + b * SS;

  const int r_ = lane & 15, hi = lane >> 4;

  const v16bf aq0 = load_frag(qp + q0 * HD, HD);
  const v16bf aq1 = load_frag(qp + q0 * HD + 32, HD);

  float mrow[8], lrow[8];
  v8f acc[4] = {};
#pragma unroll
  for (int r = 0; r < 8; ++r) { mrow[r] = NEGBIG; lrow[r] = 0.0f; }

  const int jend = q0 + 16;
  for (int j0 = 0; j0 < jend; j0 += 32) {
    // ---- batch-load the 4 K fragments, then score WMMAs (K-dim = 64) ----
    v16bf kf[4];
    kf[0] = load_frag(kp + j0 * HD, HD);
    kf[1] = load_frag(kp + j0 * HD + 32, HD);
    kf[2] = load_frag(kp + (j0 + 16) * HD, HD);
    kf[3] = load_frag(kp + (j0 + 16) * HD + 32, HD);

    // hoist vT fragment loads so they overlap the softmax VALU work
    v16bf bv[4];
#pragma unroll
    for (int t = 0; t < 4; ++t)
      bv[t] = load_frag(vp + (16 * t) * SS + j0, SS);

    v8f s0 = {}, s1 = {};
    s0 = wmma_bf16(aq0, kf[0], s0);
    s0 = wmma_bf16(aq1, kf[1], s0);
    s1 = wmma_bf16(aq0, kf[2], s1);
    s1 = wmma_bf16(aq1, kf[3], s1);

    const int c0 = j0 + r_, c1 = j0 + 16 + r_;
    const bool k0ok = mk[c0] != 0;
    const bool k1ok = mk[c1] != 0;

    float alpha[8];
#pragma unroll
    for (int r = 0; r < 8; ++r) {
      const int row = q0 + r + hi * 8;
      float v0 = s0[r] * 0.125f;               // HD^-0.5
      float v1 = s1[r] * 0.125f;
      if (c0 > row || !k0ok) v0 = NEGBIG;
      if (c1 > row || !k1ok) v1 = NEGBIG;

      float mx = fmaxf(v0, v1);                // row max across 16 lanes
      mx = fmaxf(mx, __shfl_xor(mx, 1, 32));
      mx = fmaxf(mx, __shfl_xor(mx, 2, 32));
      mx = fmaxf(mx, __shfl_xor(mx, 4, 32));
      mx = fmaxf(mx, __shfl_xor(mx, 8, 32));
      const float mn = fmaxf(mrow[r], mx);
      alpha[r] = __expf(mrow[r] - mn);
      const float p0 = __expf(v0 - mn);
      const float p1 = __expf(v1 - mn);
      s0[r] = p0; s1[r] = p1;

      float sm = p0 + p1;                      // row sum across 16 lanes
      sm += __shfl_xor(sm, 1, 32);
      sm += __shfl_xor(sm, 2, 32);
      sm += __shfl_xor(sm, 4, 32);
      sm += __shfl_xor(sm, 8, 32);
      lrow[r] = lrow[r] * alpha[r] + sm;
      mrow[r] = mn;
    }
#pragma unroll
    for (int t = 0; t < 4; ++t)
#pragma unroll
      for (int r = 0; r < 8; ++r) acc[t][r] *= alpha[r];

    // ---- C-layout P -> LDS (row-major 16xLDSP bf16), same-wave only ----
    bf16_t* pl = plds[wave];
#pragma unroll
    for (int r = 0; r < 8; ++r) {
      const int row = r + hi * 8;
      pl[row * LDSP + r_]      = (bf16_t)s0[r];
      pl[row * LDSP + 16 + r_] = (bf16_t)s1[r];
    }
    const v16bf ap = load_frag(pl, LDSP);

    // ---- acc += P(16x32) @ V(32x64) via vT fragments ----
#pragma unroll
    for (int t = 0; t < 4; ++t)
      acc[t] = wmma_bf16(ap, bv[t], acc[t]);
  }

  // ---- epilogue: normalize, zero padded-query rows, write attended bf16 ----
#pragma unroll
  for (int r = 0; r < 8; ++r) {
    const int row = q0 + r + hi * 8;
    const float inv =
        (mk[row] != 0 && lrow[r] > 0.0f) ? (1.0f / lrow[r]) : 0.0f;
#pragma unroll
    for (int t = 0; t < 4; ++t) {
      const int dcol = h * HD + 16 * t + r_;
      outp[(b * SS + row) * DD + dcol] = (bf16_t)(acc[t][r] * inv);
    }
  }
}

// ---------------------------------------------------------------------------
// Stage 3: output projection. out[4096,1024] = att @ Wout + bout (f32 out).
// ---------------------------------------------------------------------------
__global__ __launch_bounds__(256) void k_gemm_out(
    const bf16_t* __restrict__ A, const bf16_t* __restrict__ Bt,
    const float* __restrict__ bias, float* __restrict__ outp) {
  __shared__ __align__(16) bf16_t smem[2 * 64 * LDSP];
  const int wave = threadIdx.x >> 5;
  const int lane = threadIdx.x & 31;
  const int row0 = blockIdx.x * 256 + wave * 32;
  const int col0 = blockIdx.y * 64;

  v8f c[2][4] = {};
  gemm_core_256x64(A, Bt, row0, col0, smem, c);

  const int r_ = lane & 15, hi = lane >> 4;
#pragma unroll
  for (int mi = 0; mi < 2; ++mi)
#pragma unroll
    for (int t = 0; t < 4; ++t)
#pragma unroll
      for (int r = 0; r < 8; ++r) {
        const int m = row0 + 16 * mi + r + hi * 8;
        const int n = col0 + 16 * t + r_;
        outp[m * DD + n] = c[mi][t][r] + bias[n];
      }
}

// ---------------------------------------------------------------------------
// Launch
// ---------------------------------------------------------------------------
extern "C" void kernel_launch(void* const* d_in, const int* in_sizes, int n_in,
                              void* d_out, int out_size, void* d_ws, size_t ws_size,
                              hipStream_t stream) {
  const float* x            = (const float*)d_in[0];
  const unsigned char* amsk = (const unsigned char*)d_in[1];  // numpy bool
  const float* Wqkv         = (const float*)d_in[2];
  const float* bqkv         = (const float*)d_in[3];
  const float* Wout         = (const float*)d_in[4];
  const float* bout         = (const float*)d_in[5];
  float* out                = (float*)d_out;

  // Workspace layout (bf16 elements), total 48 MiB.
  bf16_t* ws    = (bf16_t*)d_ws;
  bf16_t* xb    = ws;                       // 4096*1024
  bf16_t* wqkvT = ws + 4194304;             // 3072*1024
  bf16_t* woutT = ws + 7340032;             // 1024*1024
  bf16_t* qb    = ws + 8388608;             // [B,H,S,64]
  bf16_t* kb    = ws + 12582912;            // [B,H,S,64]
  bf16_t* vtb   = ws + 16777216;            // [B,H,64,S]
  bf16_t* attb  = ws + 20971520;            // [B,S,D]

  // Stage 0: conversions / transposes
  k_cvt_bf16<<<(BS * DD + 255) / 256, 256, 0, stream>>>(x, xb, BS * DD);
  k_transpose_bf16<<<(DD * D3 + 255) / 256, 256, 0, stream>>>(Wqkv, wqkvT, DD, D3);
  k_transpose_bf16<<<(DD * DD + 255) / 256, 256, 0, stream>>>(Wout, woutT, DD, DD);

  // Stage 1: QKV projection -> q / k / vT  (block tile 256x64)
  k_gemm_qkv<<<dim3(BS / 256, D3 / 64), 256, 0, stream>>>(xb, wqkvT, bqkv, qb, kb, vtb);

  // Stage 2: flash attention (one wave per 16-query tile; 4096 tiles)
  k_attn<<<(BB * HH * (SS / 16)) / 8, 256, 0, stream>>>(qb, kb, vtb, amsk, attb);

  // Stage 3: output projection + bias -> f32 out  (block tile 256x64)
  k_gemm_out<<<dim3(BS / 256, DD / 64), 256, 0, stream>>>(attb, woutT, bout, out);
}
